// PoolingPMATopKSelf_31645319037441
// MI455X (gfx1250) — matrix operand order repaired and checked by reading, not compiled, not run on hardware
//
#include <hip/hip_runtime.h>
#include <stddef.h>

// ---------------------------------------------------------------------------
// PoolingPMATopKSelf for MI455X (gfx1250, wave32, WMMA + TDM).
// L=2, B=1, N=4096, D=256, H=8, Hd=32, TOPK=128.
// GEMMs: v_wmma_f32_16x16x32_bf16; each wave computes a 16x64 tile (4 WMMAs
// per A fragment); each 128-thread block stages its 256-column B k-chunk
// (16 KB) into LDS with the Tensor Data Mover (tensor_load_to_lds +
// s_wait_tensorcnt), then reads B fragments with ds_load_b128.
// Workspace requirement: ~104 MB (64 MB per-head score buffer reused
// sequentially across 8 heads x 2 layers on `stream`).
// ---------------------------------------------------------------------------

typedef __attribute__((ext_vector_type(16))) __bf16 v16bf;
typedef __attribute__((ext_vector_type(8)))  __bf16 v8bf;
typedef __attribute__((ext_vector_type(8)))  float  v8f;
typedef __attribute__((ext_vector_type(4)))  unsigned int u32x4;
typedef __attribute__((ext_vector_type(8)))  int i32x8;
typedef __attribute__((ext_vector_type(4)))  int i32x4;

union BfFrag { v16bf v; v8bf h[2]; };

#define SEQ_N   4096
#define DIM_D   256
#define N_HEADS 8
#define HEAD_D  32
#define TOPKK   128

static __device__ inline __bf16 f2bf(float f) {
  // round-to-nearest-even f32 -> bf16
  unsigned u = __float_as_uint(f);
  unsigned r = u + 0x7fffu + ((u >> 16) & 1u);
  unsigned short h = (unsigned short)(r >> 16);
  return __builtin_bit_cast(__bf16, h);
}

static __device__ inline unsigned fkey(float f) {
  // order-preserving float -> uint key (bigger float => bigger key)
  unsigned u = __float_as_uint(f);
  return (u & 0x80000000u) ? ~u : (u | 0x80000000u);
}

// ---------------------------------------------------------------------------
// TDM: DMA a 2D bf16 tile [rows x rowlen] (row stride = stride_elems) from
// global memory into LDS at byte offset lds_off, packed contiguously
// row-major (rowlen elems per row). D# layout per CDNA5 ISA ch.8.
// ---------------------------------------------------------------------------
static __device__ inline void tdm_load_tile_b(const __bf16* gptr, unsigned lds_off,
                                              unsigned rows, unsigned rowlen,
                                              unsigned long long stride_elems) {
  unsigned long long ga = (unsigned long long)(size_t)gptr;
  u32x4 g0;
  g0[0] = 1u;                                   // count=1, user mode, no gather
  g0[1] = lds_off;                              // lds_addr (bytes)
  g0[2] = (unsigned)ga;                         // global_addr[31:0]
  g0[3] = (unsigned)((ga >> 32) & 0x1FFFFFFu)   // global_addr[56:32]
          | (2u << 30);                         // type=2 ("image")
  i32x8 g1;
  g1[0] = 0x00010000;                           // data_size=1 (2B); mask=0
  g1[1] = (int)((rowlen & 0xffffu) << 16);      // tensor_dim0[15:0]
  g1[2] = (int)((rowlen >> 16) | ((rows & 0xffffu) << 16));   // td0 hi | td1 lo
  g1[3] = (int)((rows >> 16) | ((rowlen & 0xffffu) << 16));   // td1 hi | tile_dim0
  g1[4] = (int)(rows & 0xffffu);                // tile_dim1 ; tile_dim2=0
  g1[5] = (int)(unsigned)(stride_elems & 0xffffffffu);        // dim0_stride lo
  g1[6] = (int)(unsigned)((stride_elems >> 32) & 0xffffu);    // dim0_stride hi
  g1[7] = 0;                                    // dim1_stride unused (2D tile)
  i32x4 gz = {0, 0, 0, 0};
#if __clang_major__ >= 23
  i32x8 gz8 = {0, 0, 0, 0, 0, 0, 0, 0};
  __builtin_amdgcn_tensor_load_to_lds(g0, g1, gz, gz, gz8, 0);
#else
  __builtin_amdgcn_tensor_load_to_lds(g0, g1, gz, gz, 0);
#endif
}

// ---------------------------------------------------------------------------
// C[M,N] = A[M,K](bf16,row-major,lda) * W[N,K](bf16,row-major,ldw)^T  (f32 C)
// grid = (M/16, N/256), block = 128 (4 waves). Block output tile = 16 x 256;
// each wave owns 16 x 64 (4 accumulators, shared A fragment).
// Per 32-wide K chunk: TDM stages W[jblk..jblk+255][k0..k0+31] into LDS.
// Fragment layouts per CDNA5 ISA 7.12.2 (16-bit A 16x32, B 32x16).
// ---------------------------------------------------------------------------
__global__ __launch_bounds__(128)
void gemm_bf16_wmma(const __bf16* __restrict__ A, int lda,
                    const __bf16* __restrict__ W, int ldw,
                    float* __restrict__ C, int ldc, int K) {
  __shared__ __bf16 btile[256 * 32];            // 16 KB B k-chunk
  const int lane = threadIdx.x & 31;
  const int wave = threadIdx.x >> 5;
  const int i0   = blockIdx.x * 16;
  const int jblk = blockIdx.y * 256;
  const int half = lane >> 4;                   // lane group 0/1
  const int lm   = lane & 15;

  const __bf16* arow = A + (size_t)(i0 + lm) * lda + half * 8;
  const unsigned lds_off = (unsigned)(size_t)btile;   // LDS byte offset

  v8f acc[4] = {{}, {}, {}, {}};
  for (int k0 = 0; k0 < K; k0 += 32) {
    __syncthreads();                            // LDS safe to overwrite
    if (threadIdx.x == 0)
      tdm_load_tile_b(W + (size_t)jblk * ldw + k0, lds_off, 256, 32,
                      (unsigned long long)ldw);
    __builtin_amdgcn_s_wait_tensorcnt(0);       // issuing wave waits for DMA
    __syncthreads();                            // B tile visible to all waves

    BfFrag a;
    // A 16x32: elems 0..7 <-> K = k0 + half*8 + 0..7 ; elems 8..15 <-> +16
    a.h[0] = *(const v8bf*)(arow + k0);
    a.h[1] = *(const v8bf*)(arow + k0 + 16);
    if (k0 + 32 < K)                            // global_prefetch_b8 path
      __builtin_prefetch(arow + k0 + 32, 0, 1);

#pragma unroll
    for (int f = 0; f < 4; ++f) {
      const int colLocal = wave * 64 + f * 16 + lm;
      BfFrag b;
      // B 32x16: lane col = colLocal, elems 0..7 <-> K = k0 + half*8 + 0..7 ; elems 8..15 <-> +16
      b.h[0] = *(const v8bf*)(btile + colLocal * 32 + half * 8);
      b.h[1] = *(const v8bf*)(btile + colLocal * 32 + half * 8 + 16);
      acc[f] = __builtin_amdgcn_wmma_f32_16x16x32_bf16(
          false, a.v, false, b.v, (short)0, acc[f], false, false);
    }
  }
  // C/D layout: col = lane%16, row = (lane/16)*8 + r
#pragma unroll
  for (int f = 0; f < 4; ++f) {
    float* crow = C + (size_t)(i0 + half * 8) * ldc + jblk + wave * 64 + f * 16 + lm;
#pragma unroll
    for (int r = 0; r < 8; ++r) crow[(size_t)r * ldc] = acc[f][r];
  }
}

// ---------------------------------------------------------------------------
// elementwise f32 -> bf16 (with scale)
// ---------------------------------------------------------------------------
__global__ __launch_bounds__(256)
void cvt_f32_bf16(const float* __restrict__ src, __bf16* __restrict__ dst,
                  int n, float scale) {
  int i = blockIdx.x * 256 + threadIdx.x;
  if (i < n) dst[i] = f2bf(src[i] * scale);
}

__global__ __launch_bounds__(256)
void init_q(const float* __restrict__ x, float* __restrict__ qcur,
            __bf16* __restrict__ qbf, int n) {
  int i = blockIdx.x * 256 + threadIdx.x;
  if (i < n) { float v = x[i]; qcur[i] = v; qbf[i] = f2bf(v); }
}

__global__ __launch_bounds__(256)
void copy_out(const float* __restrict__ src, float* __restrict__ dst, int n) {
  int i = blockIdx.x * 256 + threadIdx.x;
  if (i < n) dst[i] = src[i];
}

// ---------------------------------------------------------------------------
// Per-query top-128 over 4096 scores, softmax on selected, weighted V-gather.
// One 256-thread block per query row; head `h` writes its 32 output columns.
// Exact k-th largest via 32-step radix descent on monotone uint keys.
// ---------------------------------------------------------------------------
__global__ __launch_bounds__(256)
void topk_softmax_av(const float* __restrict__ S,   // [4096,4096]
                     const float* __restrict__ V,   // [4096,256] f32
                     float* __restrict__ attn,      // [4096,256]
                     int head) {
  const int m = blockIdx.x, tid = threadIdx.x;
  const float* srow = S + (size_t)m * SEQ_N;

  float sv[16]; unsigned kv[16];
  float lmax = -3.4e38f;
#pragma unroll
  for (int i = 0; i < 16; ++i) {
    float v = srow[tid + 256 * i];
    sv[i] = v; kv[i] = fkey(v);
    lmax = fmaxf(lmax, v);
  }

  __shared__ float red[256];
  __shared__ int cnt;

  // row max
  red[tid] = lmax; __syncthreads();
  for (int s = 128; s > 0; s >>= 1) {
    if (tid < s) red[tid] = fmaxf(red[tid], red[tid + s]);
    __syncthreads();
  }
  float rmax = red[0]; __syncthreads();

  // threshold = 128th largest key
  unsigned t = 0;
  for (int b = 31; b >= 0; --b) {
    unsigned cand = t | (1u << b);
    if (tid == 0) cnt = 0;
    __syncthreads();
    int c = 0;
#pragma unroll
    for (int i = 0; i < 16; ++i) c += (kv[i] >= cand);
    if (c) atomicAdd(&cnt, c);
    __syncthreads();
    if (cnt >= TOPKK) t = cand;
    __syncthreads();
  }

  // softmax over selected + weighted V accumulation
  float wsum = 0.f, acc[HEAD_D];
#pragma unroll
  for (int d = 0; d < HEAD_D; ++d) acc[d] = 0.f;
#pragma unroll
  for (int i = 0; i < 16; ++i) {
    if (kv[i] >= t) {
      float w = __expf(sv[i] - rmax);
      wsum += w;
      const float* vr = V + (size_t)(tid + 256 * i) * DIM_D + head * HEAD_D;
#pragma unroll
      for (int d = 0; d < HEAD_D; ++d) acc[d] += w * vr[d];
    }
  }
  red[tid] = wsum; __syncthreads();
  for (int s = 128; s > 0; s >>= 1) {
    if (tid < s) red[tid] += red[tid + s];
    __syncthreads();
  }
  float inv = 1.f / red[0]; __syncthreads();

  float* orow = attn + (size_t)m * DIM_D + head * HEAD_D;
  for (int d = 0; d < HEAD_D; ++d) {
    red[tid] = acc[d]; __syncthreads();
    for (int s = 128; s > 0; s >>= 1) {
      if (tid < s) red[tid] += red[tid + s];
      __syncthreads();
    }
    if (tid == 0) orow[d] = red[0] * inv;
    __syncthreads();
  }
}

// ---------------------------------------------------------------------------
// FF top-k: keep top-128 of 256 scores per row, ReLU, emit bf16 sparse acts.
// ---------------------------------------------------------------------------
__global__ __launch_bounds__(256)
void ff_topk_act(const float* __restrict__ S, __bf16* __restrict__ actbf) {
  const int m = blockIdx.x, tid = threadIdx.x;
  float v = S[(size_t)m * DIM_D + tid];
  unsigned kv = fkey(v);
  __shared__ int cnt;
  unsigned t = 0;
  for (int b = 31; b >= 0; --b) {
    unsigned cand = t | (1u << b);
    if (tid == 0) cnt = 0;
    __syncthreads();
    if (kv >= cand) atomicAdd(&cnt, 1);
    __syncthreads();
    if (cnt >= TOPKK) t = cand;
    __syncthreads();
  }
  float a = (kv >= t) ? fmaxf(v, 0.f) : 0.f;
  actbf[(size_t)m * DIM_D + tid] = f2bf(a);
}

// ---------------------------------------------------------------------------
// q = LN(q + delta) * g + b ; also emit bf16 copy for next GEMM.
// ---------------------------------------------------------------------------
__global__ __launch_bounds__(256)
void ln_residual(float* __restrict__ qcur, const float* __restrict__ delta,
                 const float* __restrict__ g, const float* __restrict__ b,
                 __bf16* __restrict__ qbf) {
  const int m = blockIdx.x, tid = threadIdx.x;
  const size_t idx = (size_t)m * DIM_D + tid;
  float v = qcur[idx] + delta[idx];
  __shared__ float red[256];
  red[tid] = v; __syncthreads();
  for (int s = 128; s > 0; s >>= 1) {
    if (tid < s) red[tid] += red[tid + s];
    __syncthreads();
  }
  float mean = red[0] * (1.f / 256.f); __syncthreads();
  float d = v - mean;
  red[tid] = d * d; __syncthreads();
  for (int s = 128; s > 0; s >>= 1) {
    if (tid < s) red[tid] += red[tid + s];
    __syncthreads();
  }
  float var = red[0] * (1.f / 256.f);
  float y = d * rsqrtf(var + 1e-5f) * g[tid] + b[tid];
  qcur[idx] = y;
  qbf[idx] = f2bf(y);
}

// ---------------------------------------------------------------------------
extern "C" void kernel_launch(void* const* d_in, const int* in_sizes, int n_in,
                              void* d_out, int out_size, void* d_ws, size_t ws_size,
                              hipStream_t stream) {
  (void)in_sizes; (void)n_in; (void)out_size; (void)ws_size;
  const float* x  = (const float*)d_in[0];
  const float* wq = (const float*)d_in[1];
  const float* wk = (const float*)d_in[2];
  const float* wv = (const float*)d_in[3];
  const float* wo = (const float*)d_in[4];
  const float* ga = (const float*)d_in[5];
  const float* ba = (const float*)d_in[6];
  const float* wi = (const float*)d_in[7];
  const float* wf = (const float*)d_in[8];
  const float* gf = (const float*)d_in[9];
  const float* bf = (const float*)d_in[10];
  float* out = (float*)d_out;

  const int ND = SEQ_N * DIM_D;          // 1,048,576
  const int WEL = 2 * DIM_D * DIM_D;     // 131,072 per weight tensor (L=2)
  const float qscale = 0.17677669529663687f;  // 1/sqrt(32)

  // ---- carve workspace (256B aligned) ----
  char* p = (char*)d_ws;
  auto carve = [&](size_t bytes) {
    void* r = (void*)p;
    p += (bytes + 255) & ~(size_t)255;
    return r;
  };
  float*  qcur   = (float*)carve((size_t)ND * 4);
  __bf16* qbf    = (__bf16*)carve((size_t)ND * 2);
  float*  projQ  = (float*)carve((size_t)ND * 4);
  float*  projK  = (float*)carve((size_t)ND * 4);
  float*  projV  = (float*)carve((size_t)ND * 4);
  __bf16* Qbf    = (__bf16*)carve((size_t)ND * 2);
  __bf16* Kbf    = (__bf16*)carve((size_t)ND * 2);
  float*  attn   = (float*)carve((size_t)ND * 4);
  __bf16* attnbf = (__bf16*)carve((size_t)ND * 2);
  float*  tmp    = (float*)carve((size_t)ND * 4);
  __bf16* actbf  = (__bf16*)carve((size_t)ND * 2);
  __bf16* wqb = (__bf16*)carve((size_t)WEL * 2);
  __bf16* wkb = (__bf16*)carve((size_t)WEL * 2);
  __bf16* wvb = (__bf16*)carve((size_t)WEL * 2);
  __bf16* wob = (__bf16*)carve((size_t)WEL * 2);
  __bf16* wib = (__bf16*)carve((size_t)WEL * 2);
  __bf16* wfb = (__bf16*)carve((size_t)WEL * 2);
  float*  scores = (float*)carve((size_t)SEQ_N * SEQ_N * 4);   // 64 MB

  const dim3 blk256(256), blk128(128);
  const dim3 gElemW((WEL + 255) / 256), gElemA((ND + 255) / 256);
  const dim3 gRows(SEQ_N);
  const dim3 gGemmD(SEQ_N / 16, DIM_D / 256);   // N=256 GEMMs: (256,1)
  const dim3 gGemmS(SEQ_N / 16, SEQ_N / 256);   // N=4096 score GEMM: (256,16)

  // weights -> bf16 (once per launch; deterministic)
  cvt_f32_bf16<<<gElemW, blk256, 0, stream>>>(wq, wqb, WEL, 1.f);
  cvt_f32_bf16<<<gElemW, blk256, 0, stream>>>(wk, wkb, WEL, 1.f);
  cvt_f32_bf16<<<gElemW, blk256, 0, stream>>>(wv, wvb, WEL, 1.f);
  cvt_f32_bf16<<<gElemW, blk256, 0, stream>>>(wo, wob, WEL, 1.f);
  cvt_f32_bf16<<<gElemW, blk256, 0, stream>>>(wi, wib, WEL, 1.f);
  cvt_f32_bf16<<<gElemW, blk256, 0, stream>>>(wf, wfb, WEL, 1.f);
  init_q<<<gElemA, blk256, 0, stream>>>(x, qcur, qbf, ND);

  for (int l = 0; l < 2; ++l) {
    const int wo_off = l * DIM_D * DIM_D;
    // Q/K/V projections: [4096,256] = qbf * W^T
    gemm_bf16_wmma<<<gGemmD, blk128, 0, stream>>>(qbf, DIM_D, wqb + wo_off, DIM_D, projQ, DIM_D, DIM_D);
    gemm_bf16_wmma<<<gGemmD, blk128, 0, stream>>>(qbf, DIM_D, wkb + wo_off, DIM_D, projK, DIM_D, DIM_D);
    gemm_bf16_wmma<<<gGemmD, blk128, 0, stream>>>(qbf, DIM_D, wvb + wo_off, DIM_D, projV, DIM_D, DIM_D);
    // bf16 Q (with 1/sqrt(Hd) folded) and K for score WMMAs
    cvt_f32_bf16<<<gElemA, blk256, 0, stream>>>(projQ, Qbf, ND, qscale);
    cvt_f32_bf16<<<gElemA, blk256, 0, stream>>>(projK, Kbf, ND, 1.f);

    for (int h = 0; h < N_HEADS; ++h) {
      // per-head scores: [4096,4096], K=32 => one TDM chunk, 16 WMMAs/block
      gemm_bf16_wmma<<<gGemmS, blk128, 0, stream>>>(
          Qbf + h * HEAD_D, DIM_D, Kbf + h * HEAD_D, DIM_D,
          scores, SEQ_N, HEAD_D);
      topk_softmax_av<<<gRows, blk256, 0, stream>>>(scores, projV, attn, h);
    }

    // output projection + residual + LN
    cvt_f32_bf16<<<gElemA, blk256, 0, stream>>>(attn, attnbf, ND, 1.f);
    gemm_bf16_wmma<<<gGemmD, blk128, 0, stream>>>(attnbf, DIM_D, wob + wo_off, DIM_D, tmp, DIM_D, DIM_D);
    ln_residual<<<gRows, blk256, 0, stream>>>(qcur, tmp, ga + l * DIM_D, ba + l * DIM_D, qbf);

    // FF block as top-k attention over wi rows
    gemm_bf16_wmma<<<gGemmD, blk128, 0, stream>>>(qbf, DIM_D, wib + wo_off, DIM_D, tmp, DIM_D, DIM_D);
    ff_topk_act<<<gRows, blk256, 0, stream>>>(tmp, actbf);
    gemm_bf16_wmma<<<gGemmD, blk128, 0, stream>>>(actbf, DIM_D, wfb + wo_off, DIM_D, tmp, DIM_D, DIM_D);
    ln_residual<<<gRows, blk256, 0, stream>>>(qcur, tmp, gf + l * DIM_D, bf + l * DIM_D, qbf);
  }

  copy_out<<<gElemA, blk256, 0, stream>>>(qcur, out, ND);
}
